// Net_52767968199020
// MI455X (gfx1250) — compile-verified, matrix operand order checked by reading, not run on hardware
//
#include <hip/hip_runtime.h>

// ---------------------------------------------------------------------------
// PointConv-style network for MI455X (gfx1250, wave32).
// Dense layers -> v_wmma_f32_16x16x32_bf16 tiled GEMM (bf16 in, f32 acc),
// branch-free b128 tile loads via zero-padded strides + pre-transposed weights,
// double-buffered LDS with register prefetch.
// Irregular geometry (FPS / kNN / density top-100) -> LDS-cooperative kernels.
// ---------------------------------------------------------------------------

typedef __bf16 v16bf __attribute__((ext_vector_type(16)));
typedef float  v8f   __attribute__((ext_vector_type(8)));

union FragBF { unsigned int u[8]; v16bf v; };

__device__ __forceinline__ unsigned short f2bf(float f) {
    unsigned int u = __float_as_uint(f);
    u = (u + 0x7FFFu + ((u >> 16) & 1u)) >> 16;   // round-to-nearest-even
    return (unsigned short)u;
}
__device__ __forceinline__ float bf2f(unsigned short h) {
    return __uint_as_float(((unsigned int)h) << 16);
}

// ---------------------------------------------------------------------------
// bf16 GEMM:  Y[M,*] = relu(A[M,Kp] @ Wt^T + bias)   (bf16 out, stride Yp)
//   A  : [M, Kp]  row-major, zero-padded cols K..Kp-1, M % 128 == 0
//   Wt : [Np, Kp] transposed weights, zero-padded (Np = ceil128(N))
// Tile 128x128x32, 256 threads = 8 waves, wave -> 32x64 (2x4 WMMA frags).
// ---------------------------------------------------------------------------
__global__ __launch_bounds__(256)
void wmma_gemm_bias_relu(const unsigned short* __restrict__ A,
                         const unsigned short* __restrict__ Wt,
                         const float* __restrict__ bias,
                         unsigned short* __restrict__ Y,
                         int M, int Kp, int N, int Yp) {
    __shared__ unsigned short sA[2][128 * 32];
    __shared__ unsigned short sB[2][128 * 32];

    const int t    = threadIdx.x;
    const int bm   = blockIdx.y, bn = blockIdx.x;
    const int wave = t >> 5, lane = t & 31;
    const int wm   = wave >> 1, wn = wave & 1;     // 4x2 wave grid
    const int lm   = lane & 15, hi = lane >> 4;

    // Each thread moves 16 contiguous elems (32B) of one half-row of each tile
    const int e0 = t * 16, row = e0 >> 5, col0 = e0 & 31;
    const size_t aBase = (size_t)(bm * 128 + row) * Kp + col0;
    const size_t bBase = (size_t)(bn * 128 + row) * Kp + col0;

    v8f acc[2][4];
    for (int i = 0; i < 2; ++i)
        for (int j = 0; j < 4; ++j)
            for (int e = 0; e < 8; ++e) acc[i][j][e] = 0.0f;

    const int kTiles = Kp >> 5;
    uint4 ra0, ra1, rb0, rb1;
    auto gload = [&](int kt) {
        const uint4* pa = (const uint4*)(A + aBase + (size_t)kt * 32);
        ra0 = pa[0]; ra1 = pa[1];
        const uint4* pb = (const uint4*)(Wt + bBase + (size_t)kt * 32);
        rb0 = pb[0]; rb1 = pb[1];
    };
    auto sstore = [&](int buf) {
        uint4* da = (uint4*)&sA[buf][row * 32 + col0];
        da[0] = ra0; da[1] = ra1;
        uint4* db = (uint4*)&sB[buf][row * 32 + col0];
        db[0] = rb0; db[1] = rb1;
    };

    gload(0);
    sstore(0);
    if (kTiles > 1) gload(1);
    __syncthreads();

    for (int kt = 0; kt < kTiles; ++kt) {
        const int cur = kt & 1;
        FragBF af[2], bfr[4];
        for (int mi = 0; mi < 2; ++mi) {
            int r = wm * 32 + mi * 16 + lm;
            for (int j = 0; j < 8; ++j) {
                int k0 = ((j >> 2) << 4) + (hi << 3) + ((j & 3) << 1);
                af[mi].u[j] = *(const unsigned int*)&sA[cur][r * 32 + k0];
            }
        }
        for (int ni = 0; ni < 4; ++ni) {
            int cn = wn * 64 + ni * 16 + lm;
            for (int j = 0; j < 8; ++j) {
                int k0 = ((j >> 2) << 4) + (hi << 3) + ((j & 3) << 1);
                bfr[ni].u[j] = *(const unsigned int*)&sB[cur][cn * 32 + k0];
            }
        }
        for (int mi = 0; mi < 2; ++mi)
            for (int ni = 0; ni < 4; ++ni)
                acc[mi][ni] = __builtin_amdgcn_wmma_f32_16x16x32_bf16(
                    false, af[mi].v, false, bfr[ni].v,
                    (short)0, acc[mi][ni], false, false);
        if (kt + 1 < kTiles) {
            sstore(cur ^ 1);
            if (kt + 2 < kTiles) gload(kt + 2);
            __syncthreads();
        }
    }

    // Epilogue: D layout -> lane holds col = lm, rows v + hi*8.
    // Write cols < Yp so pad columns of the next layer's input are zeroed.
    for (int mi = 0; mi < 2; ++mi) {
        int row0 = bm * 128 + wm * 32 + mi * 16 + hi * 8;
        for (int ni = 0; ni < 4; ++ni) {
            int col = bn * 128 + wn * 64 + ni * 16 + lm;
            if (col >= Yp) continue;
            float bb = (col < N) ? bias[col] : 0.0f;
            for (int v = 0; v < 8; ++v) {
                int r = row0 + v;
                float val = fmaxf(acc[mi][ni][v] + bb, 0.0f);
                Y[(size_t)r * Yp + col] = f2bf(val);
            }
        }
    }
}

// ---------------------------------------------------------------------------
// Weight convert: f32 [K,N] -> bf16 transposed+padded [Np, Kp]
// ---------------------------------------------------------------------------
__global__ void convert_w_pad(const float* __restrict__ src,
                              unsigned short* __restrict__ dst,
                              int K, int N, int Kp, int Np) {
    size_t i = (size_t)blockIdx.x * 256 + threadIdx.x;
    if (i >= (size_t)Np * Kp) return;
    int n = (int)(i / Kp), k = (int)(i % Kp);
    dst[i] = (n < N && k < K) ? f2bf(src[(size_t)k * N + n]) : (unsigned short)0;
}

// ---------------------------------------------------------------------------
// BatchNorm (training-mode batch stats over M rows, biased variance)
// ---------------------------------------------------------------------------
__global__ void bn_stats(const unsigned short* __restrict__ Y, int M, int N, int Ys,
                         float* __restrict__ stats) {
    int c = blockIdx.x * 256 + threadIdx.x;
    if (c >= N) return;
    int r0 = blockIdx.y * 512;
    int r1 = r0 + 512; if (r1 > M) r1 = M;
    float s = 0.f, s2 = 0.f;
    for (int r = r0; r < r1; ++r) {
        float v = bf2f(Y[(size_t)r * Ys + c]);
        s += v; s2 += v * v;
    }
    atomicAdd(&stats[c], s);
    atomicAdd(&stats[N + c], s2);
}

__global__ void bn_finalize(const float* __restrict__ stats,
                            const float* __restrict__ g,
                            const float* __restrict__ be,
                            int M, int N, float* __restrict__ ss) {
    int c = blockIdx.x * 256 + threadIdx.x;
    if (c >= N) return;
    float mean = stats[c] / (float)M;
    float var  = stats[N + c] / (float)M - mean * mean;
    float scl  = g[c] * rsqrtf(var + 1e-5f);
    ss[c]     = scl;
    ss[N + c] = be[c] - mean * scl;
}

__global__ void bn_apply(unsigned short* __restrict__ Y,
                         const float* __restrict__ ss, int M, int N, int Ys) {
    size_t i = (size_t)blockIdx.x * 256 + threadIdx.x;
    if (i >= (size_t)M * Ys) return;
    int c = (int)(i % Ys);
    if (c >= N) return;                 // keep pad columns zero
    float v = bf2f(Y[i]) * ss[c] + ss[N + c];
    Y[i] = f2bf(v);
}

// ---------------------------------------------------------------------------
// Tiny MLP layer (Cin,Cout <= 16): per-node scalar linear + ReLU -> bf16
// ---------------------------------------------------------------------------
template <bool INF32>
__global__ void small_mlp(const void* __restrict__ X,
                          const float* __restrict__ W,
                          const float* __restrict__ Bv,
                          unsigned short* __restrict__ Y,
                          int M, int Cin, int Cout) {
    int i = blockIdx.x * 256 + threadIdx.x;
    if (i >= M) return;
    float xin[16];
    if (INF32) {
        const float* Xf = (const float*)X;
        for (int c = 0; c < Cin; ++c) xin[c] = Xf[(size_t)i * Cin + c];
    } else {
        const unsigned short* Xh = (const unsigned short*)X;
        for (int c = 0; c < Cin; ++c) xin[c] = bf2f(Xh[(size_t)i * Cin + c]);
    }
    for (int o = 0; o < Cout; ++o) {
        float a = Bv[o];
        for (int c = 0; c < Cin; ++c) a += xin[c] * W[c * Cout + o];
        Y[(size_t)i * Cout + o] = f2bf(fmaxf(a, 0.0f));
    }
}

// ---------------------------------------------------------------------------
// FPS (deterministic start at 0), writes sorted indices + sampled positions
// ---------------------------------------------------------------------------
#define MAXN 2048
__global__ void fps_kernel(const float* __restrict__ pos, int N, int S,
                           int* __restrict__ fpsIdx, float* __restrict__ spos) {
    __shared__ float px[MAXN], py[MAXN], pz[MAXN], dmin[MAXN];
    __shared__ unsigned char flag[MAXN];
    __shared__ float redV[256];
    __shared__ int   redI[256];
    __shared__ int   s_last;
    __shared__ int   cnt[256];

    int b = blockIdx.x, t = threadIdx.x;
    const float* P = pos + (size_t)b * N * 3;
    for (int i = t; i < N; i += 256) {
        px[i] = P[i * 3 + 0]; py[i] = P[i * 3 + 1]; pz[i] = P[i * 3 + 2];
        dmin[i] = 3.4e38f; flag[i] = 0;
    }
    if (t == 0) { s_last = 0; flag[0] = 1; }
    __syncthreads();

    for (int it = 1; it < S; ++it) {
        int last = s_last;
        float lx = px[last], ly = py[last], lz = pz[last];
        float bv = -1.0f; int bi = 0x7fffffff;
        for (int i = t; i < N; i += 256) {
            float dx = px[i] - lx, dy = py[i] - ly, dz = pz[i] - lz;
            float d = dx * dx + dy * dy + dz * dz;
            float dm = fminf(dmin[i], d);
            dmin[i] = dm;
            if (dm > bv || (dm == bv && i < bi)) { bv = dm; bi = i; }
        }
        redV[t] = bv; redI[t] = bi;
        __syncthreads();
        for (int off = 128; off > 0; off >>= 1) {
            if (t < off) {
                float v2 = redV[t + off]; int i2 = redI[t + off];
                if (v2 > redV[t] || (v2 == redV[t] && i2 < redI[t])) {
                    redV[t] = v2; redI[t] = i2;
                }
            }
            __syncthreads();
        }
        if (t == 0) { s_last = redI[0]; flag[redI[0]] = 1; }
        __syncthreads();
    }

    int chunk = (N + 255) / 256;
    int base = t * chunk, c = 0;
    for (int i = 0; i < chunk; ++i) {
        int idx = base + i;
        if (idx < N && flag[idx]) c++;
    }
    cnt[t] = c;
    __syncthreads();
    if (t == 0) {
        int run = 0;
        for (int i = 0; i < 256; ++i) { int v = cnt[i]; cnt[i] = run; run += v; }
    }
    __syncthreads();
    int w = cnt[t];
    for (int i = 0; i < chunk; ++i) {
        int idx = base + i;
        if (idx < N && flag[idx]) {
            fpsIdx[b * S + w] = idx;
            spos[((size_t)b * S + w) * 3 + 0] = px[idx];
            spos[((size_t)b * S + w) * 3 + 1] = py[idx];
            spos[((size_t)b * S + w) * 3 + 2] = pz[idx];
            w++;
        }
    }
}

// ---------------------------------------------------------------------------
// kNN: K nearest source points per query (cooperative K x argmin in LDS)
// ---------------------------------------------------------------------------
__global__ void knn_kernel(const float* __restrict__ q,
                           const float* __restrict__ p,
                           int N, int S, int K, int* __restrict__ nbr) {
    __shared__ float d2[MAXN];
    __shared__ float redV[256];
    __shared__ int   redI[256];
    int bs = blockIdx.x, t = threadIdx.x;
    int b = bs / S;
    const float* P = p + (size_t)b * N * 3;
    float qx = q[(size_t)bs * 3 + 0], qy = q[(size_t)bs * 3 + 1], qz = q[(size_t)bs * 3 + 2];
    for (int i = t; i < N; i += 256) {
        float dx = P[i * 3 + 0] - qx, dy = P[i * 3 + 1] - qy, dz = P[i * 3 + 2] - qz;
        d2[i] = dx * dx + dy * dy + dz * dz;
    }
    __syncthreads();
    for (int kk = 0; kk < K; ++kk) {
        float bv = 3.5e38f; int bi = 0x7fffffff;
        for (int i = t; i < N; i += 256) {
            float v = d2[i];
            if (v < bv || (v == bv && i < bi)) { bv = v; bi = i; }
        }
        redV[t] = bv; redI[t] = bi;
        __syncthreads();
        for (int off = 128; off > 0; off >>= 1) {
            if (t < off) {
                float v2 = redV[t + off]; int i2 = redI[t + off];
                if (v2 < redV[t] || (v2 == redV[t] && i2 < redI[t])) {
                    redV[t] = v2; redI[t] = i2;
                }
            }
            __syncthreads();
        }
        if (t == 0) { nbr[(size_t)bs * K + kk] = redI[0]; d2[redI[0]] = 3.6e38f; }
        __syncthreads();
    }
}

// ---------------------------------------------------------------------------
// Density: 100-NN gaussian kernel mean at sampled points -> inverse density
// ---------------------------------------------------------------------------
__global__ void density_kernel(const float* __restrict__ p,
                               const int* __restrict__ fpsIdx,
                               int N, int S, int KD, float* __restrict__ inv) {
    __shared__ float d2[MAXN];
    __shared__ float redV[256];
    __shared__ int   redI[256];
    int bs = blockIdx.x, t = threadIdx.x;
    int b = bs / S;
    int j = fpsIdx[bs];
    const float* P = p + (size_t)b * N * 3;
    float qx = P[j * 3 + 0], qy = P[j * 3 + 1], qz = P[j * 3 + 2];
    for (int i = t; i < N; i += 256) {
        float dx = P[i * 3 + 0] - qx, dy = P[i * 3 + 1] - qy, dz = P[i * 3 + 2] - qz;
        float d = dx * dx + dy * dy + dz * dz;
        d2[i] = (i == j) ? 3.6e38f : d;   // knn_graph: no self loops
    }
    __syncthreads();
    float acc = 0.0f;                      // meaningful on t==0
    const float inv2bw2 = 1.0f / (2.0f * 0.2f * 0.2f);
    const float invnrm  = 1.0f / (2.5f * 0.2f);
    for (int kk = 0; kk < KD; ++kk) {
        float bv = 3.5e38f; int bi = 0x7fffffff;
        for (int i = t; i < N; i += 256) {
            float v = d2[i];
            if (v < bv || (v == bv && i < bi)) { bv = v; bi = i; }
        }
        redV[t] = bv; redI[t] = bi;
        __syncthreads();
        for (int off = 128; off > 0; off >>= 1) {
            if (t < off) {
                float v2 = redV[t + off]; int i2 = redI[t + off];
                if (v2 < redV[t] || (v2 == redV[t] && i2 < redI[t])) {
                    redV[t] = v2; redI[t] = i2;
                }
            }
            __syncthreads();
        }
        if (t == 0) {
            float dist = sqrtf(redV[0]);
            acc += expf(-dist * inv2bw2) * invnrm;
            d2[redI[0]] = 3.6e38f;
        }
        __syncthreads();
    }
    if (t == 0) {
        float dens = acc / (float)KD;
        inv[bs] = 1.0f / (dens + 1e-10f);
    }
}

// ---------------------------------------------------------------------------
// Grouped-message builders (write padded rows, zero-filled)
// ---------------------------------------------------------------------------
__global__ void build_msg1(const float* __restrict__ pos, const float* __restrict__ x,
                           const float* __restrict__ spos, const int* __restrict__ nbr,
                           unsigned short* __restrict__ msg, unsigned short* __restrict__ grp,
                           int N, int S, int K, int Mtot, int Kp) {
    int i = blockIdx.x * 256 + threadIdx.x;
    if (i >= Mtot) return;
    int s = (i / K) % S, b = i / (K * S);
    int j = nbr[i];
    float gx = pos[((size_t)b * N + j) * 3 + 0] - spos[((size_t)b * S + s) * 3 + 0];
    float gy = pos[((size_t)b * N + j) * 3 + 1] - spos[((size_t)b * S + s) * 3 + 1];
    float gz = pos[((size_t)b * N + j) * 3 + 2] - spos[((size_t)b * S + s) * 3 + 2];
    grp[(size_t)i * 3 + 0] = f2bf(gx);
    grp[(size_t)i * 3 + 1] = f2bf(gy);
    grp[(size_t)i * 3 + 2] = f2bf(gz);
    unsigned short* row = msg + (size_t)i * Kp;
    row[0] = f2bf(gx); row[1] = f2bf(gy); row[2] = f2bf(gz);
    row[3] = f2bf(x[((size_t)b * N + j) * 3 + 0]);
    row[4] = f2bf(x[((size_t)b * N + j) * 3 + 1]);
    row[5] = f2bf(x[((size_t)b * N + j) * 3 + 2]);
    for (int c = 6; c < Kp; ++c) row[c] = 0;
}

__global__ void build_msg2(const float* __restrict__ pos1, const float* __restrict__ pos2,
                           const unsigned short* __restrict__ out1, const int* __restrict__ nbr,
                           unsigned short* __restrict__ msg, unsigned short* __restrict__ grp,
                           int N, int S, int K, int C, int Mtot, int Kp) {
    int i = blockIdx.x * 256 + threadIdx.x;
    if (i >= Mtot) return;
    int s = (i / K) % S, b = i / (K * S);
    int j = nbr[i];
    float gx = pos1[((size_t)b * N + j) * 3 + 0] - pos2[((size_t)b * S + s) * 3 + 0];
    float gy = pos1[((size_t)b * N + j) * 3 + 1] - pos2[((size_t)b * S + s) * 3 + 1];
    float gz = pos1[((size_t)b * N + j) * 3 + 2] - pos2[((size_t)b * S + s) * 3 + 2];
    grp[(size_t)i * 3 + 0] = f2bf(gx);
    grp[(size_t)i * 3 + 1] = f2bf(gy);
    grp[(size_t)i * 3 + 2] = f2bf(gz);
    unsigned short* row = msg + (size_t)i * Kp;
    row[0] = f2bf(gx); row[1] = f2bf(gy); row[2] = f2bf(gz);
    const unsigned short* src = out1 + ((size_t)b * N + j) * C;
    for (int c = 0; c < C; ++c) row[3 + c] = src[c];
    for (int c = 3 + C; c < Kp; ++c) row[c] = 0;
}

// ---------------------------------------------------------------------------
// einsum 'bskc,bskw->bscw' with per-sample density scale (W=4)
// np has padded row stride Cs; output row length C*4 (already 32-multiple)
// ---------------------------------------------------------------------------
__global__ void einsum_kernel(const unsigned short* __restrict__ np,
                              const unsigned short* __restrict__ wt,
                              const unsigned short* __restrict__ dscale,
                              unsigned short* __restrict__ out, int K, int C, int Cs) {
    int bs = blockIdx.x, c = threadIdx.x;
    if (c >= C) return;
    float a0 = 0.f, a1 = 0.f, a2 = 0.f, a3 = 0.f;
    for (int k = 0; k < K; ++k) {
        float v = bf2f(np[((size_t)bs * K + k) * Cs + c]);
        const unsigned short* w4 = wt + ((size_t)bs * K + k) * 4;
        a0 += v * bf2f(w4[0]); a1 += v * bf2f(w4[1]);
        a2 += v * bf2f(w4[2]); a3 += v * bf2f(w4[3]);
    }
    float ds = bf2f(dscale[bs]);
    unsigned short* o = out + ((size_t)bs * C + c) * 4;
    o[0] = f2bf(a0 * ds); o[1] = f2bf(a1 * ds);
    o[2] = f2bf(a2 * ds); o[3] = f2bf(a3 * ds);
}

// ---------------------------------------------------------------------------
// Tail kernels
// ---------------------------------------------------------------------------
__global__ void concat_pool(const unsigned short* __restrict__ out2,
                            const float* __restrict__ pos2,
                            unsigned short* __restrict__ A, int rows, int C2, int Kp) {
    size_t i = (size_t)blockIdx.x * 256 + threadIdx.x;
    size_t tot = (size_t)rows * Kp;
    if (i >= tot) return;
    int r = (int)(i / Kp), c = (int)(i % Kp);
    unsigned short v = 0;
    if (c < C2) v = out2[(size_t)r * C2 + c];
    else if (c < C2 + 3) v = f2bf(pos2[(size_t)r * 3 + (c - C2)]);
    A[i] = v;
}

__global__ void maxpool_kernel(const unsigned short* __restrict__ feat,
                               float* __restrict__ g, int Bn, int S, int C) {
    int i = blockIdx.x * 256 + threadIdx.x;
    if (i >= Bn * C) return;
    int b = i / C, c = i % C;
    float m = -3.4e38f;
    for (int s = 0; s < S; ++s)
        m = fmaxf(m, bf2f(feat[((size_t)b * S + s) * C + c]));
    g[i] = m;
}

__global__ void linear_kernel(const float* __restrict__ X, const float* __restrict__ W,
                              const float* __restrict__ b, float* __restrict__ Y,
                              int Bn, int K, int Nout, int relu) {
    int i = blockIdx.x * 256 + threadIdx.x;
    if (i >= Bn * Nout) return;
    int bb = i / Nout, o = i % Nout;
    float a = b[o];
    for (int k = 0; k < K; ++k) a += X[(size_t)bb * K + k] * W[(size_t)k * Nout + o];
    Y[i] = relu ? fmaxf(a, 0.0f) : a;
}

__global__ void logsoftmax_kernel(const float* __restrict__ H, float* __restrict__ out,
                                  int Bn, int C) {
    int b = threadIdx.x;
    if (b >= Bn) return;
    float m = -3.4e38f;
    for (int c = 0; c < C; ++c) m = fmaxf(m, H[b * C + c]);
    float s = 0.0f;
    for (int c = 0; c < C; ++c) s += expf(H[b * C + c] - m);
    float l = logf(s);
    for (int c = 0; c < C; ++c) out[b * C + c] = H[b * C + c] - m - l;
}

// ---------------------------------------------------------------------------
// Host orchestration
// ---------------------------------------------------------------------------
static inline int ceil32(int v)  { return (v + 31) & ~31; }
static inline int ceil128(int v) { return (v + 127) & ~127; }

extern "C" void kernel_launch(void* const* d_in, const int* in_sizes, int n_in,
                              void* d_out, int out_size, void* d_ws, size_t ws_size,
                              hipStream_t stream) {
    (void)in_sizes; (void)n_in; (void)out_size;
    const int B = 16, N = 2048, S1 = 1024, K1 = 32, S2 = 256, K2 = 64, KD = 100;
    const float* pos = (const float*)d_in[0];
    const float* x   = (const float*)d_in[1];
    auto PW = [&](int i) { return (const float*)d_in[i]; };

    // deterministic bump allocator over d_ws
    char* base = (char*)d_ws;
    size_t off = 0;
    auto alloc = [&](size_t bytes) -> void* {
        void* p = base + off;
        off = (off + bytes + 255) & ~(size_t)255;
        return p;
    };

    // transposed+padded bf16 copies of the GEMM weight matrices
    struct WC { int idx; int K; int N; unsigned short* dst; };
    WC wc[10] = {{10, 6, 16, 0},      {14, 16, 16, 0},
                 {18, 64, 256, 0},    {22, 256, 256, 0},
                 {42, 259, 256, 0},   {46, 256, 256, 0},
                 {50, 1024, 1024, 0}, {54, 1024, 2048, 0},
                 {66, 2051, 2048, 0}, {70, 2048, 2048, 0}};
    for (auto& c : wc)
        c.dst = (unsigned short*)alloc((size_t)ceil128(c.N) * ceil32(c.K) * 2);

    const size_t M1 = (size_t)B * S1 * K1;   // 524288
    const size_t M2 = (size_t)B * S2 * K2;   // 262144
    unsigned short* BUF0 = (unsigned short*)alloc((size_t)262144 * 288 * 2);
    unsigned short* BUF1 = (unsigned short*)alloc((size_t)262144 * 288 * 2);
    unsigned short* GRP  = (unsigned short*)alloc(M1 * 3 * 2);
    unsigned short* WBA  = (unsigned short*)alloc(M1 * 8 * 2);
    unsigned short* WBB  = (unsigned short*)alloc(M1 * 8 * 2);
    unsigned short* OUT1 = (unsigned short*)alloc((size_t)B * S1 * 256 * 2);
    unsigned short* OUT2 = (unsigned short*)alloc((size_t)B * S2 * 2048 * 2);
    int*   FPS1 = (int*)alloc((size_t)B * S1 * 4);
    int*   FPS2 = (int*)alloc((size_t)B * S2 * 4);
    float* POS1 = (float*)alloc((size_t)B * S1 * 3 * 4);
    float* POS2 = (float*)alloc((size_t)B * S2 * 3 * 4);
    int*   NBR1 = (int*)alloc(M1 * 4);
    int*   NBR2 = (int*)alloc(M2 * 4);
    float* DINV = (float*)alloc((size_t)B * S1 * 4);
    unsigned short* DBA = (unsigned short*)alloc((size_t)B * S1 * 8 * 2);
    unsigned short* DBB = (unsigned short*)alloc((size_t)B * S1 * 2);
    float* STATS = (float*)alloc(2 * 2048 * 4);
    float* SS    = (float*)alloc(2 * 2048 * 4);
    float* GP = (float*)alloc((size_t)B * 2048 * 4);
    float* H1 = (float*)alloc((size_t)B * 512 * 4);
    float* H2 = (float*)alloc((size_t)B * 256 * 4);
    float* H3 = (float*)alloc((size_t)B * 10 * 4);
    if (off > ws_size) return;   // workspace too small; bail deterministically

    for (auto& c : wc) {
        int Kp = ceil32(c.K), Np = ceil128(c.N);
        size_t tot = (size_t)Np * Kp;
        convert_w_pad<<<(int)((tot + 255) / 256), 256, 0, stream>>>(
            (const float*)d_in[c.idx], c.dst, c.K, c.N, Kp, Np);
    }

    // A stride = ceil32(K); Yp = padded output stride (next layer's K stride)
    auto gemm_layer = [&](const unsigned short* A, int M, int Kc, int Nc, int pb,
                          const unsigned short* Wb, unsigned short* Yb, int Yp) {
        int Kp = ceil32(Kc), Np = ceil128(Nc);
        hipMemsetAsync(STATS, 0, (size_t)2 * Nc * 4, stream);
        dim3 g(Np / 128, M / 128);
        wmma_gemm_bias_relu<<<g, 256, 0, stream>>>(A, Wb, PW(pb + 1), Yb, M, Kp, Nc, Yp);
        dim3 gs((Nc + 255) / 256, (M + 511) / 512);
        bn_stats<<<gs, 256, 0, stream>>>(Yb, M, Nc, Yp, STATS);
        bn_finalize<<<(Nc + 255) / 256, 256, 0, stream>>>(STATS, PW(pb + 2), PW(pb + 3), M, Nc, SS);
        size_t tot = (size_t)M * Yp;
        bn_apply<<<(int)((tot + 255) / 256), 256, 0, stream>>>(Yb, SS, M, Nc, Yp);
    };
    auto small_layer = [&](const void* X, bool inF32, int M, int Cin, int Cout, int pb,
                           unsigned short* Yb) {
        hipMemsetAsync(STATS, 0, (size_t)2 * Cout * 4, stream);
        int gb = (M + 255) / 256;
        if (inF32) small_mlp<true><<<gb, 256, 0, stream>>>(X, PW(pb), PW(pb + 1), Yb, M, Cin, Cout);
        else       small_mlp<false><<<gb, 256, 0, stream>>>(X, PW(pb), PW(pb + 1), Yb, M, Cin, Cout);
        dim3 gs(1, (M + 511) / 512);
        bn_stats<<<gs, 256, 0, stream>>>(Yb, M, Cout, Cout, STATS);
        bn_finalize<<<1, 256, 0, stream>>>(STATS, PW(pb + 2), PW(pb + 3), M, Cout, SS);
        size_t tot = (size_t)M * Cout;
        bn_apply<<<(int)((tot + 255) / 256), 256, 0, stream>>>(Yb, SS, M, Cout, Cout);
    };

    // ---- set abstraction 1 ----
    fps_kernel<<<B, 256, 0, stream>>>(pos, N, S1, FPS1, POS1);
    knn_kernel<<<B * S1, 256, 0, stream>>>(POS1, pos, N, S1, K1, NBR1);
    density_kernel<<<B * S1, 256, 0, stream>>>(pos, FPS1, N, S1, KD, DINV);
    small_layer(DINV, true, B * S1, 1, 8, 26, DBA);
    small_layer(DBA, false, B * S1, 8, 1, 30, DBB);
    build_msg1<<<(int)((M1 + 255) / 256), 256, 0, stream>>>(pos, x, POS1, NBR1, BUF0, GRP,
                                                            N, S1, K1, (int)M1, 32);
    small_layer(GRP, false, (int)M1, 3, 8, 2, WBA);
    small_layer(WBA, false, (int)M1, 8, 4, 6, WBB);
    gemm_layer(BUF0, (int)M1, 6, 16, 10, wc[0].dst, BUF1, 32);
    gemm_layer(BUF1, (int)M1, 16, 16, 14, wc[1].dst, BUF0, 32);
    einsum_kernel<<<B * S1, 64, 0, stream>>>(BUF0, WBB, DBB, BUF1, K1, 16, 32);
    gemm_layer(BUF1, B * S1, 64, 256, 18, wc[2].dst, BUF0, 256);
    gemm_layer(BUF0, B * S1, 256, 256, 22, wc[3].dst, OUT1, 256);

    // ---- set abstraction 2 ----
    fps_kernel<<<B, 256, 0, stream>>>(POS1, S1, S2, FPS2, POS2);
    knn_kernel<<<B * S2, 256, 0, stream>>>(POS2, POS1, S1, S2, K2, NBR2);
    density_kernel<<<B * S2, 256, 0, stream>>>(POS1, FPS2, S1, S2, KD, DINV);
    small_layer(DINV, true, B * S2, 1, 8, 58, DBA);
    small_layer(DBA, false, B * S2, 8, 1, 62, DBB);
    build_msg2<<<(int)((M2 + 255) / 256), 256, 0, stream>>>(POS1, POS2, OUT1, NBR2, BUF0, GRP,
                                                            S1, S2, K2, 256, (int)M2, 288);
    small_layer(GRP, false, (int)M2, 3, 8, 34, WBA);
    small_layer(WBA, false, (int)M2, 8, 4, 38, WBB);
    gemm_layer(BUF0, (int)M2, 259, 256, 42, wc[4].dst, BUF1, 256);
    gemm_layer(BUF1, (int)M2, 256, 256, 46, wc[5].dst, BUF0, 256);
    einsum_kernel<<<B * S2, 256, 0, stream>>>(BUF0, WBB, DBB, BUF1, K2, 256, 256);
    gemm_layer(BUF1, B * S2, 1024, 1024, 50, wc[6].dst, BUF0, 1024);
    gemm_layer(BUF0, B * S2, 1024, 2048, 54, wc[7].dst, OUT2, 2048);

    // ---- pool MLP + head ----
    size_t cc = (size_t)B * S2 * 2080;
    concat_pool<<<(int)((cc + 255) / 256), 256, 0, stream>>>(OUT2, POS2, BUF0, B * S2, 2048, 2080);
    gemm_layer(BUF0, B * S2, 2051, 2048, 66, wc[8].dst, BUF1, 2048);
    gemm_layer(BUF1, B * S2, 2048, 2048, 70, wc[9].dst, BUF0, 2048);
    maxpool_kernel<<<(B * 2048 + 255) / 256, 256, 0, stream>>>(BUF0, GP, B, S2, 2048);
    linear_kernel<<<(B * 512 + 255) / 256, 256, 0, stream>>>(GP, PW(74), PW(75), H1, B, 2048, 512, 1);
    linear_kernel<<<(B * 256 + 255) / 256, 256, 0, stream>>>(H1, PW(76), PW(77), H2, B, 512, 256, 1);
    linear_kernel<<<(B * 10 + 255) / 256, 256, 0, stream>>>(H2, PW(78), PW(79), H3, B, 256, 10, 0);
    logsoftmax_kernel<<<1, 32, 0, stream>>>(H3, (float*)d_out, B, 10);
}
